// WindowAttention_38809324487100
// MI455X (gfx1250) — matrix operand order repaired and checked by reading, not compiled, not run on hardware
//
#include <hip/hip_runtime.h>
#include <hip/hip_bf16.h>

typedef __attribute__((ext_vector_type(16))) _Float16 v16h;
typedef __attribute__((ext_vector_type(8)))  _Float16 v8h;
typedef __attribute__((ext_vector_type(8)))  float    v8f;

#define DIM   128
#define NTOK  49
#define NPAD  64
#define HEADS 4
#define HD    32

// ---- WMMA helpers -----------------------------------------------------------

__device__ __forceinline__ v8f wmma_f16(v16h a, v16h b, v8f c) {
  // 8 args: (neg_a, A, neg_b, B, c_mod, C, reuse_a, reuse_b)
  return __builtin_amdgcn_wmma_f32_16x16x32_f16(false, a, false, b, (short)0, c,
                                                false, false);
}

// A-matrix 16x32 f16 fragment from a row-major matrix (per ISA 7.12.2):
// lane l (m = l&15, g = l>>4): halves 0..7 = row[k0 + g*8 .. +7],
//                              halves 8..15 = row[k0 + 16 + g*8 .. +7]
__device__ __forceinline__ v16h load_a_frag(const _Float16* base, int stride,
                                            int row0, int k0, int lane) {
  int m = lane & 15, g = lane >> 4;
  const _Float16* p = base + (row0 + m) * stride + k0 + g * 8;
  union { v16h v; v8h h[2]; } u;
  u.h[0] = *(const v8h*)(p);
  u.h[1] = *(const v8h*)(p + 16);
  return u.v;
}

// B-matrix 32x16 f16 fragment where B[k][n] = W[n][k] (W row-major):
// lane l (n = n0 + (l&15), g = l>>4): halves 0..7 = W[n][k0 + g*8 .. +7],
//                                     halves 8..15 = W[n][k0 + 16 + g*8 .. +7]
__device__ __forceinline__ v16h load_b_frag_rm(const _Float16* W, int stride,
                                               int k0, int n0, int lane) {
  int n = lane & 15, g = lane >> 4;
  const _Float16* p = W + (n0 + n) * stride + k0 + g * 8;
  union { v16h v; v8h h[2]; } u;
  u.h[0] = *(const v8h*)(p);
  u.h[1] = *(const v8h*)(p + 16);
  return u.v;
}

// ---- Prep: weights fp32 -> f16 in scratch, bias table -> padded [4][64][64] --

__global__ __launch_bounds__(256) void prep_kernel(
    const float* __restrict__ qw, const float* __restrict__ kw,
    const float* __restrict__ vw, const float* __restrict__ pw,
    const float* __restrict__ bias_table, const int* __restrict__ rel_idx,
    _Float16* __restrict__ w16, float* __restrict__ biasM) {
  int idx = blockIdx.x * 256 + threadIdx.x;
  if (idx < 4 * DIM * DIM) {
    int which = idx >> 14, e = idx & 16383;
    const float* src = (which == 0) ? qw : (which == 1) ? kw
                     : (which == 2) ? vw : pw;
    w16[idx] = (_Float16)src[e];
  }
  if (idx < HEADS * NPAD * NPAD) {
    int h = idx >> 12, rem = idx & 4095, n = rem >> 6, m = rem & 63;
    float v = 0.f;
    if (n < NTOK && m < NTOK)
      v = bias_table[rel_idx[n * NTOK + m] * HEADS + h];
    biasM[idx] = v;
  }
}

// ---- Main: one workgroup (8 wave32) per window ------------------------------

__global__ __launch_bounds__(256) void win_attn_kernel(
    const float* __restrict__ x,
    const float* __restrict__ q_b, const float* __restrict__ k_b,
    const float* __restrict__ v_b, const float* __restrict__ p_b,
    const _Float16* __restrict__ w16,   // [4][128][128] f16: q,k,v,proj
    const float* __restrict__ biasM,    // [4][64][64] f32
    float* __restrict__ out) {
  __shared__ _Float16 Xs[NPAD * DIM];   // 16 KB  x (f16, padded rows = 0)
  __shared__ _Float16 Qs[NPAD * DIM];   // 16 KB  q * scale
  __shared__ _Float16 Ks[NPAD * DIM];   // 16 KB
  __shared__ _Float16 Vt[DIM * NPAD];   // 16 KB  v transposed [d][m]
  __shared__ float    Sb[NPAD * 68];    // 17 KB  logits (stride 68 vs conflicts)
  __shared__ _Float16 Ps[NPAD * NPAD];  //  8 KB  softmax probs
  __shared__ _Float16 Os[NPAD * DIM];   // 16 KB  attention output

  const int tid  = threadIdx.x;
  const int lane = tid & 31;
  const int wave = tid >> 5;
  const long long b = blockIdx.x;
  const float scale = 0.17677669529663687f;  // 32^-0.5

  // ---- Phase 1: stage x[b] into LDS as f16, zero-pad rows 49..63
  const float* xb = x + b * (NTOK * DIM);
  for (int i = tid; i < NPAD * (DIM / 4); i += 256) {
    int row = i >> 5, c4 = (i & 31) << 2;
    float4 v = make_float4(0.f, 0.f, 0.f, 0.f);
    if (row < NTOK) v = *(const float4*)(xb + row * DIM + c4);
    _Float16* d = Xs + row * DIM + c4;
    d[0] = (_Float16)v.x; d[1] = (_Float16)v.y;
    d[2] = (_Float16)v.z; d[3] = (_Float16)v.w;
  }
  __syncthreads();

  // ---- Phase 2: Q,K,V = x @ W.T + b   (96 tiles of 16x16, 12 per wave)
  for (int i = 0; i < 12; ++i) {
    int t = wave + (i << 3);
    int mat = t >> 5;                 // 0=Q 1=K 2=V
    int r5 = t & 31;
    int mi = r5 >> 3, ni = r5 & 7;
    const _Float16* W = w16 + mat * (DIM * DIM);
    v8f acc = {};
    for (int ks = 0; ks < 4; ++ks) {
      v16h a  = load_a_frag(Xs, DIM, mi * 16, ks * 32, lane);
      v16h bb = load_b_frag_rm(W, DIM, ks * 32, ni * 16, lane);
      acc = wmma_f16(a, bb, acc);
    }
    int col  = ni * 16 + (lane & 15);
    int row0 = mi * 16 + ((lane >> 4) << 3);
    if (mat == 0) {
      float bias = q_b[col];
      for (int r = 0; r < 8; ++r)
        Qs[(row0 + r) * DIM + col] = (_Float16)((acc[r] + bias) * scale);
    } else if (mat == 1) {
      float bias = k_b[col];
      for (int r = 0; r < 8; ++r)
        Ks[(row0 + r) * DIM + col] = (_Float16)(acc[r] + bias);
    } else {
      float bias = v_b[col];
      for (int r = 0; r < 8; ++r)
        Vt[col * NPAD + row0 + r] = (_Float16)(acc[r] + bias);
    }
  }
  __syncthreads();

  // ---- Phase 3: per-head attention
  for (int h = 0; h < HEADS; ++h) {
    // S = (q*scale) @ k^T + bias : 16 tiles, 2 per wave, K-dim = 32 (1 WMMA)
    for (int i = 0; i < 2; ++i) {
      int t = wave + (i << 3);
      int ti = t >> 2, tj = t & 3;
      v16h a  = load_a_frag(Qs, DIM, ti * 16, h * HD, lane);
      v16h bb = load_b_frag_rm(Ks, DIM, h * HD, tj * 16, lane);
      v8f acc = {};
      acc = wmma_f16(a, bb, acc);
      int col  = tj * 16 + (lane & 15);
      int row0 = ti * 16 + ((lane >> 4) << 3);
      const float* bp = biasM + (h * NPAD + row0) * NPAD + col;
      for (int r = 0; r < 8; ++r)
        Sb[(row0 + r) * 68 + col] = acc[r] + bp[r * NPAD];
    }
    __syncthreads();

    // Softmax over valid cols 0..48, one row per thread (threads 0..63)
    if (tid < NPAD) {
      int row = tid;
      float mx = -1e30f;
      for (int m = 0; m < NTOK; ++m) mx = fmaxf(mx, Sb[row * 68 + m]);
      float sum = 0.f;
      for (int m = 0; m < NTOK; ++m) sum += __expf(Sb[row * 68 + m] - mx);
      float inv = 1.0f / sum;
      for (int m = 0; m < NTOK; ++m)
        Ps[row * NPAD + m] = (_Float16)(__expf(Sb[row * 68 + m] - mx) * inv);
      for (int m = NTOK; m < NPAD; ++m) Ps[row * NPAD + m] = (_Float16)0.f;
    }
    __syncthreads();

    // O_head = P @ V_head : 8 tiles, 1 per wave, K-dim = 64 (2 WMMAs)
    {
      int ti = wave >> 1, tdj = wave & 1;
      v8f acc = {};
      for (int ks = 0; ks < 2; ++ks) {
        v16h a  = load_a_frag(Ps, NPAD, ti * 16, ks * 32, lane);
        v16h bb = load_b_frag_rm(Vt, NPAD, ks * 32, h * HD + tdj * 16, lane);
        acc = wmma_f16(a, bb, acc);
      }
      int col  = h * HD + tdj * 16 + (lane & 15);
      int row0 = ti * 16 + ((lane >> 4) << 3);
      for (int r = 0; r < 8; ++r)
        Os[(row0 + r) * DIM + col] = (_Float16)acc[r];
    }
    __syncthreads();  // Sb/Ps reuse next head; Os complete for proj
  }

  // ---- Phase 4: out = O @ proj_w.T + proj_b : 32 tiles, 4 per wave
  const _Float16* Wp = w16 + 3 * (DIM * DIM);
  for (int i = 0; i < 4; ++i) {
    int t = wave + (i << 3);
    int mi = t >> 3, ni = t & 7;
    v8f acc = {};
    for (int ks = 0; ks < 4; ++ks) {
      v16h a  = load_a_frag(Os, DIM, mi * 16, ks * 32, lane);
      v16h bb = load_b_frag_rm(Wp, DIM, ks * 32, ni * 16, lane);
      acc = wmma_f16(a, bb, acc);
    }
    int col  = ni * 16 + (lane & 15);
    int row0 = mi * 16 + ((lane >> 4) << 3);
    float bias = p_b[col];
    for (int r = 0; r < 8; ++r) {
      int n = row0 + r;
      if (n < NTOK)
        out[b * (NTOK * DIM) + n * DIM + col] = acc[r] + bias;
    }
  }
}

// ---- Launch -----------------------------------------------------------------

extern "C" void kernel_launch(void* const* d_in, const int* in_sizes, int n_in,
                              void* d_out, int out_size, void* d_ws,
                              size_t ws_size, hipStream_t stream) {
  const float* x  = (const float*)d_in[0];
  const float* qw = (const float*)d_in[1];
  const float* qb = (const float*)d_in[2];
  const float* kw = (const float*)d_in[3];
  const float* kb = (const float*)d_in[4];
  const float* vw = (const float*)d_in[5];
  const float* vb = (const float*)d_in[6];
  const float* pw = (const float*)d_in[7];
  const float* pb = (const float*)d_in[8];
  const float* bt = (const float*)d_in[9];
  const int* ridx = (const int*)d_in[10];
  float* out = (float*)d_out;

  _Float16* w16 = (_Float16*)d_ws;                               // 128 KB
  float* biasM = (float*)((char*)d_ws + 4 * DIM * DIM * sizeof(_Float16));

  prep_kernel<<<256, 256, 0, stream>>>(qw, kw, vw, pw, bt, ridx, w16, biasM);
  win_attn_kernel<<<16384, 256, 0, stream>>>(x, qb, kb, vb, pb, w16, biasM, out);
}
